// CausalAttention_78065325572367
// MI455X (gfx1250) — compile-verified, matrix operand order checked by reading, not run on hardware
//
#include <hip/hip_runtime.h>
#include <hip/hip_bf16.h>
#include <cstdint>
#include <cstddef>

typedef __attribute__((ext_vector_type(16))) __bf16 v16bf;
typedef __attribute__((ext_vector_type(8)))  float  v8f;
typedef __attribute__((ext_vector_type(4)))  unsigned int v4u;
typedef __attribute__((ext_vector_type(8)))  int v8i;
typedef __attribute__((ext_vector_type(4)))  int v4i;

#define DEVINL __device__ __forceinline__

#if defined(__gfx1250__) && __has_builtin(__builtin_amdgcn_tensor_load_to_lds)
#define HAVE_TDM 1
#else
#define HAVE_TDM 0
#endif

static constexpr int Tt = 4096, Bb = 8, Ee = 1024, Hh = 8, Dd = 128, Kk = 64;
static constexpr int Mrows = Tt * Bb;   // 32768
static constexpr int NC = Tt / 128;     // 32 chunks

DEVINL v8f wmma_bf16(v16bf a, v16bf b, v8f c) {
  // D = A(16x32 bf16) * B(32x16 bf16) + C(16x16 f32)
  return __builtin_amdgcn_wmma_f32_16x16x32_bf16(false, a, false, b, (short)0, c,
                                                 false, false);
}

DEVINL v8f vzero8() {
  v8f z;
#pragma unroll
  for (int i = 0; i < 8; ++i) z[i] = 0.f;
  return z;
}

// ---------------------------------------------------------------------------
// fp32 -> bf16 conversion (vectorized by 4)
// ---------------------------------------------------------------------------
__global__ __launch_bounds__(256) void cvt_kernel(const float* __restrict__ in,
                                                  __bf16* __restrict__ out,
                                                  size_t n) {
  size_t i = ((size_t)blockIdx.x * 256 + threadIdx.x) * 4;
  if (i + 3 < n) {
    float4 v = *(const float4*)(in + i);
    out[i + 0] = (__bf16)v.x;
    out[i + 1] = (__bf16)v.y;
    out[i + 2] = (__bf16)v.z;
    out[i + 3] = (__bf16)v.w;
  }
}

// ---------------------------------------------------------------------------
// C[M,N] = A[M,K] @ W[N,K]^T + bias   (A,W bf16; C bf16 or f32)
// 128x128 block tile, 256 threads (8 waves), each wave owns a 16x128 strip.
// K is staged 32 wide through padded LDS; WMMA bf16 -> f32 accumulate.
// ---------------------------------------------------------------------------
template <bool BF16OUT>
__global__ __launch_bounds__(256) void gemm_bias_kernel(
    const __bf16* __restrict__ A, const __bf16* __restrict__ W,
    const float* __restrict__ bias, void* __restrict__ Cout, int Kdim, int N) {
  constexpr int LDT = 48;  // 32 + 16 pad (96B rows: 32B-aligned fragments)
  __shared__ __align__(32) __bf16 As[128 * LDT];
  __shared__ __align__(32) __bf16 Bs[128 * LDT];

  const int tid = threadIdx.x, lane = tid & 31, wave = tid >> 5;
  const int lhi = lane >> 4, llo = lane & 15;
  const int n0 = blockIdx.x * 128, m0 = blockIdx.y * 128;

  v8f acc[8];
#pragma unroll
  for (int j = 0; j < 8; ++j) acc[j] = vzero8();

  for (int kk = 0; kk < Kdim; kk += 32) {
    // cooperative stage: 128 rows x 32 cols of A and W
    for (int s = tid; s < 128 * 4; s += 256) {
      int r = s >> 2, cs = (s & 3) * 8;
      *(uint4*)&As[r * LDT + cs] =
          *(const uint4*)(A + (size_t)(m0 + r) * Kdim + kk + cs);
      *(uint4*)&Bs[r * LDT + cs] =
          *(const uint4*)(W + (size_t)(n0 + r) * Kdim + kk + cs);
    }
    if (kk + 32 < Kdim) {  // global_prefetch_b8 of next K tile
      __builtin_prefetch(A + (size_t)(m0 + (tid >> 1)) * Kdim + kk + 32, 0, 0);
      __builtin_prefetch(W + (size_t)(n0 + (tid >> 1)) * Kdim + kk + 32, 0, 0);
    }
    __syncthreads();

    v16bf af = *(const v16bf*)&As[(wave * 16 + llo) * LDT + lhi * 16];
#pragma unroll
    for (int j = 0; j < 8; ++j) {
      v16bf bf = *(const v16bf*)&Bs[(j * 16 + llo) * LDT + lhi * 16];
      acc[j] = wmma_bf16(af, bf, acc[j]);
    }
    __syncthreads();
  }

  // epilogue: C/D layout -> row = base + g + 8*(lane>>4), col = tile + (lane&15)
#pragma unroll
  for (int j = 0; j < 8; ++j) {
    int n = n0 + j * 16 + llo;
    float bv = bias[n];
    int mb = m0 + wave * 16 + lhi * 8;
#pragma unroll
    for (int g = 0; g < 8; ++g) {
      float val = acc[j][g] + bv;
      size_t off = (size_t)(mb + g) * N + n;
      if (BF16OUT)
        ((__bf16*)Cout)[off] = (__bf16)val;
      else
        ((float*)Cout)[off] = val;
    }
  }
}

// ---------------------------------------------------------------------------
// phi(u)[r,h, k]      = sin(p) * K^-1/2
// phi(u)[r,h, K + k]  = cos(p) * K^-1/2
// p = D^-0.25 * sum_d u[r, h*D+d] * sigma[h,d] * rm[h,k,d]   (TAU = 1)
// ---------------------------------------------------------------------------
__global__ __launch_bounds__(256) void phi_kernel(
    const __bf16* __restrict__ u, const float* __restrict__ rmx,
    const float* __restrict__ sigma, __bf16* __restrict__ out) {
  const float qs = 0.29730177875068026f;  // 128^-0.25
  const float os = 0.125f;                // 64^-0.5
  size_t t = (size_t)blockIdx.x * 256 + threadIdx.x;
  int k = (int)(t & 63);
  size_t rh = t >> 6;
  int h = (int)(rh & 7);
  size_t r = rh >> 3;  // t*B + b
  const __bf16* up = u + r * (size_t)Ee + h * 128;
  const float* rp = rmx + ((size_t)h * Kk + k) * Dd;
  const float* sp = sigma + (size_t)h * Dd;
  float p = 0.f;
#pragma unroll 4
  for (int d = 0; d < 128; ++d) p += (float)up[d] * (rp[d] * sp[d]);
  p *= qs;
  float sn, cs;
  __sincosf(p, &sn, &cs);
  __bf16* op = out + r * (size_t)Ee + h * 128;
  op[k] = (__bf16)(sn * os);
  op[64 + k] = (__bf16)(cs * os);
}

// ---------------------------------------------------------------------------
// Chunked causal linear attention, one block per (b,h), 32 sequential chunks.
// Per chunk: scores = Qc Kc^T (masked), num = scores Vc + Qc S, den = rowsum +
// Qc z, out = num/max(den,eps); then S += Kc^T Vc, z += sum Kc.
// S (128x128 f32) and all bf16 tiles live in dynamic LDS (~222 KB).
// Chunk tiles are staged with the Tensor Data Mover when available: one wave
// issues three 2D TDM descriptors (row stride 16 KB in HBM, pad_enable adds
// 8 DWORDs every 64 DWORDs to produce the 288 B padded LDS rows), then waits
// on TENSORcnt before the workgroup barrier.
// ---------------------------------------------------------------------------
__global__ __launch_bounds__(256) void rfa_kernel(const __bf16* __restrict__ pq,
                                                  const __bf16* __restrict__ pk,
                                                  const __bf16* __restrict__ vb,
                                                  __bf16* __restrict__ attn) {
  constexpr int LDP = 144;  // 128 + 16 pad, 288B rows (32B aligned)
  extern __shared__ __align__(32) char smem[];
  __bf16* sq = (__bf16*)smem;
  __bf16* sk = sq + 128 * LDP;
  __bf16* sv = sk + 128 * LDP;
  __bf16* ssc = sv + 128 * LDP;
  float* sS = (float*)(ssc + 128 * LDP);
  float* sz = sS + 128 * LDP;
  float* sden = sz + 128;

  const int tid = threadIdx.x, lane = tid & 31, wave = tid >> 5;
  const int lhi = lane >> 4, llo = lane & 15;
  const int b = blockIdx.x >> 3, h = blockIdx.x & 7;
  const int rb = wave * 16;  // this wave's 16-row strip

  for (int i = tid; i < 128 * LDP; i += 256) sS[i] = 0.f;
  if (tid < 128) sz[tid] = 0.f;
  __syncthreads();

  for (int c = 0; c < NC; ++c) {
    // ---- stage Qc, Kc, Vc tiles (128 x 128 bf16 each) ----
#if HAVE_TDM
    if (tid < 32) {  // one wave issues the three tensor DMAs
      const unsigned long long gbase0 = (unsigned long long)(uintptr_t)pq;
      const unsigned long long gbase1 = (unsigned long long)(uintptr_t)pk;
      const unsigned long long gbase2 = (unsigned long long)(uintptr_t)vb;
      const unsigned long long rowoff =
          (((unsigned long long)(c * 128) * Bb + b) * (unsigned long long)Ee +
           (unsigned long long)(h * 128)) * 2ull;
#pragma unroll
      for (int m = 0; m < 3; ++m) {
        unsigned long long ga =
            (m == 0 ? gbase0 : (m == 1 ? gbase1 : gbase2)) + rowoff;
        unsigned ldsoff = (unsigned)(m * 128 * LDP * 2);  // 0 / 36864 / 73728
        // D# group 0: count=1, lds_addr, global_addr[56:0], type=2
        v4u g0 = {1u, ldsoff, (unsigned)(ga & 0xffffffffu),
                  (unsigned)((ga >> 32) & 0x1ffffffu) | (2u << 30)};
        // D# group 1: data_size=2B, pad_enable, pad_interval=64DW (code 5),
        // pad_amount=8DW (code 7); tensor_dim0/1 = 128, tile_dim0/1 = 128,
        // tensor_dim0_stride = 8192 elements (B*E row stride).
        v8i g1 = {(int)((1u << 16) | (1u << 20) | (5u << 22) | (7u << 25)),
                  (int)(128u << 16),  // tensor_dim0 lo -> [31:16]
                  (int)(128u << 16),  // tensor_dim0 hi=0 | tensor_dim1 lo
                  (int)(128u << 16),  // tensor_dim1 hi=0 | tile_dim0
                  128,                // tile_dim1 | tile_dim2=0
                  8192,               // tensor_dim0_stride lo
                  0, 0};              // stride0 hi | stride1
        v4i g2 = {0, 0, 0, 0};
        v4i g3 = {0, 0, 0, 0};
#if __clang_major__ >= 23
        v8i g4 = {0, 0, 0, 0, 0, 0, 0, 0};
        __builtin_amdgcn_tensor_load_to_lds(g0, g1, g2, g3, g4, 0);
#else
        __builtin_amdgcn_tensor_load_to_lds(g0, g1, g2, g3, 0);
#endif
      }
      __builtin_amdgcn_s_wait_tensorcnt(0);
    }
#else
    for (int s = tid; s < 128 * 16; s += 256) {
      int i = s >> 4, cs = (s & 15) * 8;
      size_t g = ((size_t)(c * 128 + i) * Bb + b) * (size_t)Ee + h * 128 + cs;
      *(uint4*)&sq[i * LDP + cs] = *(const uint4*)&pq[g];
      *(uint4*)&sk[i * LDP + cs] = *(const uint4*)&pk[g];
      *(uint4*)&sv[i * LDP + cs] = *(const uint4*)&vb[g];
    }
#endif
    __syncthreads();

    // phase 1: intra-chunk scores = Qc Kc^T, causal mask, stash as bf16
    for (int j = 0; j < 8; ++j) {
      v8f acc = vzero8();
      const __bf16* ap = &sq[(rb + llo) * LDP + lhi * 16];
      const __bf16* bp = &sk[(j * 16 + llo) * LDP + lhi * 16];
#pragma unroll
      for (int kk = 0; kk < 128; kk += 32)
        acc = wmma_bf16(*(const v16bf*)(ap + kk), *(const v16bf*)(bp + kk), acc);
      int ibase = rb + lhi * 8, sidx = j * 16 + llo;
#pragma unroll
      for (int g = 0; g < 8; ++g) {
        int row = ibase + g;
        float val = (sidx <= row) ? acc[g] : 0.f;
        ssc[row * LDP + sidx] = (__bf16)val;
      }
    }
    __syncthreads();

    // denominator: row-sum of scores + q . z  (z from previous chunks)
    if (tid < 128) {
      float den = 0.f;
      for (int s2 = 0; s2 < 128; ++s2) den += (float)ssc[tid * LDP + s2];
      for (int k2 = 0; k2 < 128; ++k2)
        den += (float)sq[tid * LDP + k2] * sz[k2];
      sden[tid] = fmaxf(den, 1e-6f);
    }
    __syncthreads();

    // phase 2: num = scores @ Vc + Qc @ S ; normalize ; write out
    for (int dj = 0; dj < 8; ++dj) {
      v8f acc = vzero8();
      const int col = dj * 16 + llo;
      const __bf16* ap = &ssc[(rb + llo) * LDP + lhi * 16];
      const __bf16* aq = &sq[(rb + llo) * LDP + lhi * 16];
#pragma unroll
      for (int ss = 0; ss < 128; ss += 32) {
        v16bf bf;
        const __bf16* vp = &sv[(ss + lhi * 16) * LDP + col];
#pragma unroll
        for (int u = 0; u < 16; ++u) bf[u] = vp[u * LDP];
        acc = wmma_bf16(*(const v16bf*)(ap + ss), bf, acc);
      }
#pragma unroll
      for (int kk = 0; kk < 128; kk += 32) {
        v16bf bf;
        const float* spp = &sS[(kk + lhi * 16) * LDP + col];
#pragma unroll
        for (int u = 0; u < 16; ++u) bf[u] = (__bf16)spp[u * LDP];
        acc = wmma_bf16(*(const v16bf*)(aq + kk), bf, acc);
      }
      int ibase = rb + lhi * 8;
#pragma unroll
      for (int g = 0; g < 8; ++g) {
        int row = ibase + g;
        float val = acc[g] / sden[row];
        size_t go =
            ((size_t)(c * 128 + row) * Bb + b) * (size_t)Ee + h * 128 + col;
        attn[go] = (__bf16)val;
      }
    }
    __syncthreads();

    // phase 3: S += Kc^T Vc (this wave owns 16 k-rows), z += column sums of Kc
    for (int dj = 0; dj < 8; ++dj) {
      const int col = dj * 16 + llo;
      const int kb = rb + lhi * 8;
      v8f acc;
#pragma unroll
      for (int g = 0; g < 8; ++g) acc[g] = sS[(kb + g) * LDP + col];
#pragma unroll
      for (int ss = 0; ss < 128; ss += 32) {
        v16bf af, bf;
        const __bf16* kp = &sk[(ss + lhi * 16) * LDP + rb + llo];
        const __bf16* vp = &sv[(ss + lhi * 16) * LDP + col];
#pragma unroll
        for (int u = 0; u < 16; ++u) {
          af[u] = kp[u * LDP];
          bf[u] = vp[u * LDP];
        }
        acc = wmma_bf16(af, bf, acc);
      }
#pragma unroll
      for (int g = 0; g < 8; ++g) sS[(kb + g) * LDP + col] = acc[g];
    }
    if (tid < 128) {
      float zs = 0.f;
      for (int i = 0; i < 128; ++i) zs += (float)sk[i * LDP + tid];
      sz[tid] += zs;
    }
    __syncthreads();
  }
}

// ---------------------------------------------------------------------------
// Launch
// ---------------------------------------------------------------------------
extern "C" void kernel_launch(void* const* d_in, const int* in_sizes, int n_in,
                              void* d_out, int out_size, void* d_ws,
                              size_t ws_size, hipStream_t stream) {
  (void)in_sizes; (void)n_in; (void)out_size; (void)ws_size;

  const float* x = (const float*)d_in[0];
  const float* rm = (const float*)d_in[1];
  const float* Wq = (const float*)d_in[2];
  const float* bq = (const float*)d_in[3];
  const float* Wk = (const float*)d_in[4];
  const float* bk = (const float*)d_in[5];
  const float* Wv = (const float*)d_in[6];
  const float* bv = (const float*)d_in[7];
  const float* Wo = (const float*)d_in[8];
  const float* bo = (const float*)d_in[9];
  const float* sigma = (const float*)d_in[10];

  char* ws = (char*)d_ws;
  const size_t SZ_ACT = (size_t)Mrows * Ee * sizeof(__bf16);  // 64 MiB
  const size_t SZ_W = (size_t)Ee * Ee * sizeof(__bf16);       // 2 MiB
  __bf16* xb = (__bf16*)(ws);
  __bf16* qb = (__bf16*)(ws + 1 * SZ_ACT);
  __bf16* kb = (__bf16*)(ws + 2 * SZ_ACT);
  __bf16* vbuf = (__bf16*)(ws + 3 * SZ_ACT);
  __bf16* pq = (__bf16*)(ws + 4 * SZ_ACT);
  __bf16* pk = (__bf16*)(ws + 5 * SZ_ACT);
  __bf16* ab = (__bf16*)(ws + 6 * SZ_ACT);
  __bf16* Wqb = (__bf16*)(ws + 7 * SZ_ACT + 0 * SZ_W);
  __bf16* Wkb = (__bf16*)(ws + 7 * SZ_ACT + 1 * SZ_W);
  __bf16* Wvb = (__bf16*)(ws + 7 * SZ_ACT + 2 * SZ_W);
  __bf16* Wob = (__bf16*)(ws + 7 * SZ_ACT + 3 * SZ_W);

  const size_t nAct = (size_t)Mrows * Ee;  // 33,554,432
  const size_t nW = (size_t)Ee * Ee;       // 1,048,576
  cvt_kernel<<<(unsigned)(nAct / 4 / 256), 256, 0, stream>>>(x, xb, nAct);
  cvt_kernel<<<(unsigned)(nW / 4 / 256), 256, 0, stream>>>(Wq, Wqb, nW);
  cvt_kernel<<<(unsigned)(nW / 4 / 256), 256, 0, stream>>>(Wk, Wkb, nW);
  cvt_kernel<<<(unsigned)(nW / 4 / 256), 256, 0, stream>>>(Wv, Wvb, nW);
  cvt_kernel<<<(unsigned)(nW / 4 / 256), 256, 0, stream>>>(Wo, Wob, nW);

  dim3 gg(Ee / 128, Mrows / 128);  // (8, 256)
  gemm_bias_kernel<true><<<gg, 256, 0, stream>>>(xb, Wqb, bq, qb, Ee, Ee);
  gemm_bias_kernel<true><<<gg, 256, 0, stream>>>(xb, Wkb, bk, kb, Ee, Ee);
  gemm_bias_kernel<true><<<gg, 256, 0, stream>>>(xb, Wvb, bv, vbuf, Ee, Ee);

  const size_t nphi = (size_t)Mrows * Hh * Kk;  // 16,777,216
  phi_kernel<<<(unsigned)(nphi / 256), 256, 0, stream>>>(qb, rm, sigma, pq);
  phi_kernel<<<(unsigned)(nphi / 256), 256, 0, stream>>>(kb, rm, sigma, pk);

  constexpr int rfaLds = 4 * 128 * 144 * 2 + 128 * 144 * 4 + 2 * 128 * 4;
  (void)hipFuncSetAttribute(reinterpret_cast<const void*>(rfa_kernel),
                            hipFuncAttributeMaxDynamicSharedMemorySize, rfaLds);
  rfa_kernel<<<Bb * Hh, 256, rfaLds, stream>>>(pq, pk, vbuf, ab);

  gemm_bias_kernel<false><<<gg, 256, 0, stream>>>(ab, Wob, bo, d_out, Ee, Ee);
}
